// KernelAttention_4922032521488
// MI455X (gfx1250) — matrix operand order repaired and checked by reading, not compiled, not run on hardware
//
#include <hip/hip_runtime.h>
#include <hip/hip_bf16.h>

typedef __attribute__((ext_vector_type(4)))  float    v4f;
typedef __attribute__((ext_vector_type(8)))  float    v8f;
typedef __attribute__((ext_vector_type(16))) _Float16 v16h;

#define BM 128
#define BN 128
#define BK 32
#define LSTR 40   // padded LDS row stride in f16 elements (80B rows, keeps 16B alignment)
#define TILE_ELEMS (BM * LSTR)          // elements per LDS buffer
#define TILE_BYTES (TILE_ELEMS * 2)     // 10240 bytes

// async 16-byte global -> LDS copy (CDNA5 GLOBAL_LOAD_ASYNC_TO_LDS_B128, ASYNCcnt)
__device__ __forceinline__ void async_copy16(unsigned lds_byte_addr, const void* gptr) {
  asm volatile("global_load_async_to_lds_b128 %0, %1, off"
               :: "v"(lds_byte_addr), "v"(gptr)
               : "memory");
}

// ---------------- fp32 -> f16 conversion ----------------
__global__ __launch_bounds__(256) void cvt_f32_to_f16(const float* __restrict__ in,
                                                      _Float16* __restrict__ out,
                                                      long long n) {
  long long i = (long long)blockIdx.x * blockDim.x + threadIdx.x;
  long long stride = (long long)gridDim.x * blockDim.x;
  for (; i < n; i += stride) out[i] = (_Float16)in[i];
}

// ---------------- generic WMMA GEMM: C = A @ Bw^T (+bias, epilogue modes) ----
// A : [M x K] f16 row-major (lda), batched by strideA
// Bw: [N x K] f16 row-major (ldb), batched by strideB  (i.e. Linear weight layout)
// mode 0: store f16 C[m*ldc + n]
// mode 1: store f16 transposed C[n*ldc + m]          (used for v^T)
// mode 2: store f32 C[m*ldc + n] (+bias)             (final output)
// mode 3: store f32 theta(val)                        (attention scores)
__global__ __launch_bounds__(256) void gemm_wmma_f16(
    const _Float16* __restrict__ A, const _Float16* __restrict__ Bw,
    void* __restrict__ Cv, const float* __restrict__ bias,
    int M, int N, int K, int lda, int ldb, int ldc,
    long long strideA, long long strideB, long long strideC, int mode)
{
  __shared__ alignas(16) _Float16 smA[2 * TILE_ELEMS];  // double buffered
  __shared__ alignas(16) _Float16 smB[2 * TILE_ELEMS];

  const int b = blockIdx.z;
  A  += strideA * b;
  Bw += strideB * b;

  const int tid  = threadIdx.x;
  const int lane = tid & 31;
  const int wid  = tid >> 5;          // 8 waves
  const int wm   = wid & 1;           // 2 waves along M (64 rows each)
  const int wn   = wid >> 1;          // 4 waves along N (32 cols each)
  const int blockM = blockIdx.y * BM;
  const int blockN = blockIdx.x * BN;
  const int h  = lane >> 4;           // half-wave select per ISA fragment layout
  const int ml = lane & 15;

  v8f acc[4][2];
#pragma unroll
  for (int mt = 0; mt < 4; ++mt)
#pragma unroll
    for (int nt = 0; nt < 2; ++nt)
      acc[mt][nt] = (v8f){0.f,0.f,0.f,0.f,0.f,0.f,0.f,0.f};

  // global->LDS mapping: thread moves 32 bytes = half a 32-wide row (two b128 ops)
  const int lrow  = tid >> 1;         // 0..127
  const int lhalf = tid & 1;          // which 16-element half
  const unsigned loffB = (unsigned)(lrow * LSTR + lhalf * 16) * 2u;  // byte offset in tile
  const _Float16* gA = A  + (long long)(blockM + lrow) * lda + lhalf * 16;
  const _Float16* gB = Bw + (long long)(blockN + lrow) * ldb + lhalf * 16;

  // low 32 bits of a generic shared pointer == LDS byte address (flat aperture rule)
  const unsigned ldsA = (unsigned)(unsigned long long)(void*)smA;
  const unsigned ldsB = (unsigned)(unsigned long long)(void*)smB;

  auto issue_tile = [&](int kt, int buf) {
    const _Float16* ga = gA + kt * BK;
    const _Float16* gb = gB + kt * BK;
    const unsigned la = ldsA + (unsigned)buf * TILE_BYTES + loffB;
    const unsigned lb = ldsB + (unsigned)buf * TILE_BYTES + loffB;
    async_copy16(la,      ga);
    async_copy16(la + 16, ga + 8);
    async_copy16(lb,      gb);
    async_copy16(lb + 16, gb + 8);
  };

  auto compute_tile = [&](int buf) {
    const _Float16* bufA = smA + buf * TILE_ELEMS;
    const _Float16* bufB = smB + buf * TILE_ELEMS;
    union Frag { v16h v; v4f f[2]; } af[4], bf[2];
    // A fragment (16x32): lane(m=ml, half h) holds K = h*8..h*8+7 and 16+h*8..+7
#pragma unroll
    for (int mt = 0; mt < 4; ++mt) {
      const _Float16* p = bufA + (wm * 64 + mt * 16 + ml) * LSTR;
      af[mt].f[0] = *(const v4f*)(p + h * 8);
      af[mt].f[1] = *(const v4f*)(p + 16 + h * 8);
    }
    // B fragment (32x16): lane(n=ml, half h) holds K = h*8..h*8+7 and 16+h*8..+7 of row n
#pragma unroll
    for (int nt = 0; nt < 2; ++nt) {
      const _Float16* p = bufB + (wn * 32 + nt * 16 + ml) * LSTR + h * 8;
      bf[nt].f[0] = *(const v4f*)(p);
      bf[nt].f[1] = *(const v4f*)(p + 16);
    }
#pragma unroll
    for (int mt = 0; mt < 4; ++mt)
#pragma unroll
      for (int nt = 0; nt < 2; ++nt)
        acc[mt][nt] = __builtin_amdgcn_wmma_f32_16x16x32_f16(
            false, af[mt].v, false, bf[nt].v, (short)0, acc[mt][nt], false, false);
  };

  const int nk = K / BK;
  issue_tile(0, 0);

  // steady state: branch-free body; next tile streams in while current computes
  for (int kt = 0; kt < nk - 1; ++kt) {
    const int cur = kt & 1;
    issue_tile(kt + 1, cur ^ 1);
    // async ops complete in order: <=4 outstanding -> current tile resident
    asm volatile("s_wait_asynccnt 0x4" ::: "memory");
    __syncthreads();
    compute_tile(cur);
    __syncthreads();  // all waves done reading buf before next overwrite
  }
  // peeled last tile
  asm volatile("s_wait_asynccnt 0x0" ::: "memory");
  __syncthreads();
  compute_tile((nk - 1) & 1);

  // Epilogue: C/D fragment layout: lane(n=ml, half h), vgpr r -> (m = 8h + r, n)
#pragma unroll
  for (int mt = 0; mt < 4; ++mt)
#pragma unroll
    for (int nt = 0; nt < 2; ++nt) {
      const int gm0 = blockM + wm * 64 + mt * 16 + h * 8;
      const int gn  = blockN + wn * 32 + nt * 16 + ml;
      const float bval = bias ? bias[gn] : 0.0f;
#pragma unroll
      for (int r = 0; r < 8; ++r) {
        const int gm = gm0 + r;
        const float val = acc[mt][nt][r] + bval;
        if (mode == 0) {
          ((_Float16*)Cv + strideC * b)[(long long)gm * ldc + gn] = (_Float16)val;
        } else if (mode == 1) {
          ((_Float16*)Cv + strideC * b)[(long long)gn * ldc + gm] = (_Float16)val;
        } else if (mode == 2) {
          ((float*)Cv + strideC * b)[(long long)gm * ldc + gn] = val;
        } else {
          const float t = 0.5f
                        + 0.2f  / (1.0f + __expf(-val))
                        + 0.15f * tanhf(val)
                        + 0.1f  * fmaxf(val, 0.0f);
          ((float*)Cv + strideC * b)[(long long)gm * ldc + gn] = t;
        }
      }
    }
}

// ---------------- row softmax: theta(f32) -> P(f16), in place over theta ----
__global__ __launch_bounds__(256) void softmax_row_f16(const float* __restrict__ theta,
                                                       _Float16* __restrict__ P,
                                                       int S, long long pStride)
{
  const long long row = blockIdx.x;
  const float* tr = theta + row * S;      // S == 2048, 8 elems/thread
  _Float16* pr = P + row * pStride;
  const int tid = threadIdx.x;

  float v[8];
  float mx = -3.0e38f;
#pragma unroll
  for (int i = 0; i < 8; ++i) { v[i] = tr[tid + i * 256]; mx = fmaxf(mx, v[i]); }

  __shared__ float red[256];
  red[tid] = mx; __syncthreads();
  for (int s = 128; s > 0; s >>= 1) {
    if (tid < s) red[tid] = fmaxf(red[tid], red[tid + s]);
    __syncthreads();
  }
  mx = red[0]; __syncthreads();

  float sum = 0.f;
#pragma unroll
  for (int i = 0; i < 8; ++i) { v[i] = __expf(v[i] - mx); sum += v[i]; }
  red[tid] = sum; __syncthreads();
  for (int s = 128; s > 0; s >>= 1) {
    if (tid < s) red[tid] += red[tid + s];
    __syncthreads();
  }
  const float inv = 1.0f / red[0];
#pragma unroll
  for (int i = 0; i < 8; ++i) pr[tid + i * 256] = (_Float16)(v[i] * inv);
}

extern "C" void kernel_launch(void* const* d_in, const int* in_sizes, int n_in,
                              void* d_out, int out_size, void* d_ws, size_t ws_size,
                              hipStream_t stream)
{
  (void)in_sizes; (void)n_in; (void)out_size; (void)ws_size;
  const int B = 2, S = 2048, D = 1024;
  const long long SD = (long long)S * D;
  const long long SS = (long long)S * S;

  const float* x  = (const float*)d_in[0];
  const float* Wq = (const float*)d_in[1];
  const float* bq = (const float*)d_in[2];
  const float* Wk = (const float*)d_in[3];
  const float* bk = (const float*)d_in[4];
  const float* Wv = (const float*)d_in[5];
  const float* bv = (const float*)d_in[6];
  const float* Wo = (const float*)d_in[7];
  const float* bo = (const float*)d_in[8];

  char* ws = (char*)d_ws;
  size_t off = 0;
  auto alloc = [&](size_t bytes) -> void* {
    void* p = ws + off;
    off = (off + bytes + 255) & ~(size_t)255;
    return p;
  };
  _Float16* xh   = (_Float16*)alloc((size_t)B * SD * 2);
  _Float16* wqh  = (_Float16*)alloc((size_t)D * D * 2);
  _Float16* wkh  = (_Float16*)alloc((size_t)D * D * 2);
  _Float16* wvh  = (_Float16*)alloc((size_t)D * D * 2);
  _Float16* woh  = (_Float16*)alloc((size_t)D * D * 2);
  _Float16* qh   = (_Float16*)alloc((size_t)B * SD * 2);
  _Float16* kh   = (_Float16*)alloc((size_t)B * SD * 2);
  _Float16* vTh  = (_Float16*)alloc((size_t)B * SD * 2);   // [B][D][S]
  float*    theta = (float*)alloc((size_t)B * SS * 4);     // fp32 scores; P(f16) overlaid
  _Float16* aoh  = xh;                                     // attn output reuses x buffer

  dim3 blk(256);
  cvt_f32_to_f16<<<512, blk, 0, stream>>>(x,  xh,  (long long)B * SD);
  cvt_f32_to_f16<<<256, blk, 0, stream>>>(Wq, wqh, (long long)D * D);
  cvt_f32_to_f16<<<256, blk, 0, stream>>>(Wk, wkh, (long long)D * D);
  cvt_f32_to_f16<<<256, blk, 0, stream>>>(Wv, wvh, (long long)D * D);
  cvt_f32_to_f16<<<256, blk, 0, stream>>>(Wo, woh, (long long)D * D);

  dim3 gProj(D / BN, S / BM, B);    // (8,16,2)
  dim3 gScor(S / BN, S / BM, B);    // (16,16,2)

  // q = x @ Wq^T + bq  (f16 out)
  gemm_wmma_f16<<<gProj, blk, 0, stream>>>(xh, wqh, qh, bq,
      S, D, D, D, D, D, SD, 0, SD, 0);
  // k = x @ Wk^T + bk
  gemm_wmma_f16<<<gProj, blk, 0, stream>>>(xh, wkh, kh, bk,
      S, D, D, D, D, D, SD, 0, SD, 0);
  // v^T: store (x @ Wv^T + bv) transposed as [D][S]
  gemm_wmma_f16<<<gProj, blk, 0, stream>>>(xh, wvh, vTh, bv,
      S, D, D, D, D, S, SD, 0, SD, 1);
  // theta = f(q @ k^T)  (fp32)
  gemm_wmma_f16<<<gScor, blk, 0, stream>>>(qh, kh, theta, nullptr,
      S, S, D, D, D, S, SD, SD, SS, 3);
  // row softmax -> P (f16, packed into first half of each theta row)
  softmax_row_f16<<<B * S, blk, 0, stream>>>(theta, (_Float16*)theta, S, 2LL * S);
  // out = P @ v  == P @ (vT)^T   (f16 out)
  gemm_wmma_f16<<<gProj, blk, 0, stream>>>((const _Float16*)theta, vTh, aoh, nullptr,
      S, D, S, 2 * S, S, D, 2LL * SS, SD, SD, 0);
  // result = out @ Wo^T + bo  (fp32 to d_out)
  gemm_wmma_f16<<<gProj, blk, 0, stream>>>(aoh, woh, d_out, bo,
      S, D, D, D, D, D, SD, 0, SD, 2);
}
